// PointNetSetAbstraction_70643622084670
// MI455X (gfx1250) — matrix operand order repaired and checked by reading, not compiled, not run on hardware
//
#include <hip/hip_runtime.h>

// ---------------------------------------------------------------------------
// PointNet Set Abstraction for MI455X (gfx1250, wave32, WMMA f16->f32)
// B=16, N=2048, S=N/2+1=1025, k=16, EMB=64, mlp 66->64->128->256
// ---------------------------------------------------------------------------

typedef _Float16 v16h __attribute__((ext_vector_type(16)));
typedef float    v8f  __attribute__((ext_vector_type(8)));

#define BB   16
#define NN   2048
#define SS   1025
#define KNBR 16
#define EMBD 64
#define PCOL (BB * SS * KNBR)   // 262400 columns, multiple of 16

// ---------------------------------------------------------------- transpose
__global__ void k_transpose_xyz(const float* __restrict__ xy,
                                float* __restrict__ xyz) {
  int t = blockIdx.x * blockDim.x + threadIdx.x;
  if (t >= BB * NN) return;
  int b = t / NN, n = t % NN;
  xyz[2 * t]     = xy[(size_t)(b * 2 + 0) * NN + n];
  xyz[2 * t + 1] = xy[(size_t)(b * 2 + 1) * NN + n];
}

// ---------------------------------------------------------------- FPS
__global__ void k_fps(const float* __restrict__ xyz, int* __restrict__ fps_idx) {
  __shared__ float sv[256];
  __shared__ int   si[256];
  const int b = blockIdx.x;
  const int t = threadIdx.x;
  const float* xb = xyz + (size_t)b * NN * 2;
  const int PPT = NN / 256;  // 8 points per thread
  float px[PPT], py[PPT], dist[PPT];
#pragma unroll
  for (int i = 0; i < PPT; ++i) {
    int n = i * 256 + t;
    px[i] = xb[2 * n];
    py[i] = xb[2 * n + 1];
    dist[i] = 3.4e38f;
  }
  int far = 0;
  for (int s = 0; s < SS; ++s) {
    if (t == 0) fps_idx[b * SS + s] = far;   // record BEFORE update (matches scan)
    float cx = xb[2 * far], cy = xb[2 * far + 1];
    float bv = -1.0f;
    int   bi = 0;
#pragma unroll
    for (int i = 0; i < PPT; ++i) {
      float dx = px[i] - cx, dy = py[i] - cy;
      float dd = dx * dx + dy * dy;
      dist[i] = fminf(dist[i], dd);
      if (dist[i] > bv) { bv = dist[i]; bi = i * 256 + t; }
    }
    sv[t] = bv; si[t] = bi;
    __syncthreads();
    for (int off = 128; off > 0; off >>= 1) {
      if (t < off) {
        float ov = sv[t + off]; int oi = si[t + off];
        if (ov > sv[t] || (ov == sv[t] && oi < si[t])) { sv[t] = ov; si[t] = oi; }
      }
      __syncthreads();
    }
    far = si[0];
    __syncthreads();
  }
}

// ------------------------------------------------- gather centers (+ output0)
__global__ void k_new_xyz(const float* __restrict__ xyz,
                          const int* __restrict__ fps_idx,
                          float* __restrict__ new_xyz,
                          float* __restrict__ out0) {
  int t = blockIdx.x * blockDim.x + threadIdx.x;
  if (t >= BB * SS) return;
  int b = t / SS, s = t % SS;
  int j = fps_idx[t];
  float x = xyz[((size_t)b * NN + j) * 2];
  float y = xyz[((size_t)b * NN + j) * 2 + 1];
  new_xyz[2 * t]     = x;
  new_xyz[2 * t + 1] = y;
  out0[(size_t)(b * 2 + 0) * SS + s] = x;   // new_xyz transposed [B,2,S]
  out0[(size_t)(b * 2 + 1) * SS + s] = y;
}

// ---------------------------------------------------------------- KNN (k=16)
__global__ void k_knn(const float* __restrict__ xyz, int* __restrict__ knn) {
  int t = blockIdx.x * blockDim.x + threadIdx.x;
  if (t >= BB * NN) return;
  int b = t / NN, n = t % NN;
  const float* xb = xyz + (size_t)b * NN * 2;
  float qx = xb[2 * n], qy = xb[2 * n + 1];
  float dl[KNBR];
  int   il[KNBR];
#pragma unroll
  for (int j = 0; j < KNBR; ++j) { dl[j] = 3.4e38f; il[j] = 0; }
  for (int m = 0; m < NN; ++m) {
    float dx = xb[2 * m] - qx, dy = xb[2 * m + 1] - qy;
    float d = dx * dx + dy * dy;
    if (d < dl[KNBR - 1]) {
      dl[KNBR - 1] = d; il[KNBR - 1] = m;
#pragma unroll
      for (int j = KNBR - 1; j > 0; --j) {
        if (dl[j] < dl[j - 1]) {
          float td = dl[j]; dl[j] = dl[j - 1]; dl[j - 1] = td;
          int   ti = il[j]; il[j] = il[j - 1]; il[j - 1] = ti;
        }
      }
    }
  }
#pragma unroll
  for (int j = 0; j < KNBR; ++j) knn[(size_t)t * KNBR + j] = il[j];
}

// ------------------------------------------- build X0 [P][96] f16 (K0 pad 96)
__global__ void k_build_x0(const float* __restrict__ xyz,
                           const float* __restrict__ fea,
                           const int* __restrict__ fps_idx,
                           const int* __restrict__ knn,
                           const float* __restrict__ new_xyz,
                           _Float16* __restrict__ X0) {
  int t = blockIdx.x * blockDim.x + threadIdx.x;  // column index p
  if (t >= PCOL) return;
  int kk = t & 15;
  int bs = t >> 4;
  int b  = bs / SS;
  int center = fps_idx[bs];
  int j = knn[((size_t)b * NN + center) * KNBR + kk];
  _Float16* xc = X0 + (size_t)t * 96;
  float nx = new_xyz[2 * bs], ny = new_xyz[2 * bs + 1];
  xc[0] = (_Float16)(xyz[((size_t)b * NN + j) * 2]     - nx);
  xc[1] = (_Float16)(xyz[((size_t)b * NN + j) * 2 + 1] - ny);
  const float* fb = fea + (size_t)b * EMBD * NN;
#pragma unroll
  for (int c = 0; c < EMBD; ++c) xc[2 + c] = (_Float16)fb[(size_t)c * NN + j];
#pragma unroll
  for (int c = 2 + EMBD; c < 96; ++c) xc[c] = (_Float16)0.0f;
}

// -------------------------------------------------- pack weights f32 -> f16
__global__ void k_pack_w(const float* __restrict__ W, _Float16* __restrict__ Wp,
                         int M, int K, int Kp) {
  int t = blockIdx.x * blockDim.x + threadIdx.x;
  if (t >= M * Kp) return;
  int m = t / Kp, k = t % Kp;
  Wp[t] = (_Float16)((k < K) ? W[(size_t)m * K + k] : 0.0f);
}

// -------------------- GEMM: Y[M][P] = W[M][K] * X[P][K]^T + b ; BN stats ----
// One wave computes one 16x16 tile of Y via V_WMMA_F32_16X16X32_F16.
__global__ void k_gemm_wmma(const _Float16* __restrict__ Wp,
                            const _Float16* __restrict__ X,
                            const float* __restrict__ bias,
                            _Float16* __restrict__ Y,
                            float* __restrict__ bsum,
                            float* __restrict__ bsumsq,
                            int M, int K, int P) {
  int wave = (blockIdx.x * blockDim.x + threadIdx.x) >> 5;
  int lane = threadIdx.x & 31;
  int mtiles = M >> 4;
  int ntiles = P >> 4;
  if (wave >= mtiles * ntiles) return;     // wave-uniform exit (EXEC all-1s in WMMA)
  int mt = wave % mtiles;
  int nt = wave / mtiles;
  int hf  = lane >> 4;                     // lane half selects K sub-block
  int lid = lane & 15;
  const _Float16* wrow = Wp + (size_t)((mt << 4) + lid) * K;  // A row m
  const _Float16* xrow = X  + (size_t)((nt << 4) + lid) * K;  // B column n
  v8f acc = {};
  for (int kc = 0; kc < K; kc += 32) {
    v16h a, b;
#pragma unroll
    for (int i = 0; i < 4; ++i) {          // A 16x32: VGPR0-3 K=hf*8+2i, VGPR4-7 K=16+hf*8+2i
      int ka = kc + hf * 8 + 2 * i;
      a[2 * i]     = wrow[ka];
      a[2 * i + 1] = wrow[ka + 1];
      int kb = kc + 16 + hf * 8 + 2 * i;
      a[2 * i + 8] = wrow[kb];
      a[2 * i + 9] = wrow[kb + 1];
    }
#pragma unroll
    for (int i = 0; i < 8; ++i) {          // B 32x16: same K order as A (VGPR0-3 K=hf*8+2i, VGPR4-7 K=16+hf*8+2(i-4))
      int kb = kc + hf * 8 + 2 * i + (i >= 4 ? 8 : 0);
      b[2 * i]     = xrow[kb];
      b[2 * i + 1] = xrow[kb + 1];
    }
    acc = __builtin_amdgcn_wmma_f32_16x16x32_f16(false, a, false, b,
                                                 (short)0, acc, false, false);
  }
  // Epilogue: bias, store f16 Y, wave-level BN partial sums -> atomics.
  int col = (nt << 4) + lid;
#pragma unroll
  for (int r = 0; r < 8; ++r) {            // C/D: VGPR r -> row r (+8 for upper half)
    int row = (mt << 4) + hf * 8 + r;
    float v = acc[r] + bias[row];
    Y[(size_t)row * P + col] = (_Float16)v;
    float s = v, s2 = v * v;
#pragma unroll
    for (int off = 8; off > 0; off >>= 1) { // reduce across the 16 lanes of this row
      s  += __shfl_down(s,  off, 16);
      s2 += __shfl_down(s2, off, 16);
    }
    if (lid == 0) {
      atomicAdd(&bsum[row],   s);
      atomicAdd(&bsumsq[row], s2);
    }
  }
}

// ------------------------------------------------ fold BN stats into scale/shift
__global__ void k_bn_finalize(const float* __restrict__ bsum,
                              const float* __restrict__ bsumsq,
                              const float* __restrict__ g,
                              const float* __restrict__ be,
                              float* __restrict__ scale,
                              float* __restrict__ shift,
                              int M, float invP) {
  int m = blockIdx.x * blockDim.x + threadIdx.x;
  if (m >= M) return;
  float mean = bsum[m] * invP;
  float var  = bsumsq[m] * invP - mean * mean;
  float rstd = rsqrtf(var + 1e-5f);
  float sc = g[m] * rstd;
  scale[m] = sc;
  shift[m] = be[m] - mean * sc;
}

// ----------------------- BN + ReLU + transpose to next-layer layout [P][M] f16
__global__ void k_bn_relu_tr(const _Float16* __restrict__ Y,
                             const float* __restrict__ scale,
                             const float* __restrict__ shift,
                             _Float16* __restrict__ Xn,
                             int M, int P) {
  size_t t = (size_t)blockIdx.x * blockDim.x + threadIdx.x;
  if (t >= (size_t)M * P) return;
  int    m = (int)(t / P);
  size_t p = t - (size_t)m * P;
  float v = (float)Y[t] * scale[m] + shift[m];
  v = fmaxf(v, 0.0f);
  Xn[p * M + m] = (_Float16)v;
}

// -------------------- final layer: BN + ReLU + max over k -> out [B,256,S] f32
__global__ void k_bn_relu_maxk(const _Float16* __restrict__ Y,
                               const float* __restrict__ scale,
                               const float* __restrict__ shift,
                               float* __restrict__ out2,
                               int Mo, int P) {
  int t = blockIdx.x * blockDim.x + threadIdx.x;
  int total = BB * Mo * SS;
  if (t >= total) return;
  int s  = t % SS;
  int bc = t / SS;
  int c  = bc % Mo;
  int b  = bc / Mo;
  size_t colbase = ((size_t)b * SS + s) * 16;
  const _Float16* yr = Y + (size_t)c * P + colbase;
  float sc = scale[c], sh = shift[c];
  float mx = 0.0f;                          // relu => all candidates >= 0
#pragma unroll
  for (int kk = 0; kk < 16; ++kk) {
    float v = (float)yr[kk] * sc + sh;
    v = fmaxf(v, 0.0f);
    mx = fmaxf(mx, v);
  }
  out2[t] = mx;                             // t == (b*Mo + c)*SS + s
}

// ---------------------------------------------------------------------------
extern "C" void kernel_launch(void* const* d_in, const int* in_sizes, int n_in,
                              void* d_out, int out_size, void* d_ws, size_t ws_size,
                              hipStream_t stream) {
  (void)in_sizes; (void)n_in; (void)out_size; (void)ws_size;
  const float* xy  = (const float*)d_in[0];
  const float* fea = (const float*)d_in[1];
  const float* W1  = (const float*)d_in[2];
  const float* b1  = (const float*)d_in[3];
  const float* g1  = (const float*)d_in[4];
  const float* be1 = (const float*)d_in[5];
  const float* W2  = (const float*)d_in[6];
  const float* b2  = (const float*)d_in[7];
  const float* g2  = (const float*)d_in[8];
  const float* be2 = (const float*)d_in[9];
  const float* W3  = (const float*)d_in[10];
  const float* b3  = (const float*)d_in[11];
  const float* g3  = (const float*)d_in[12];
  const float* be3 = (const float*)d_in[13];

  const int P = PCOL;                       // 262400
  char* base = (char*)d_ws;
  size_t off = 0;
  auto alloc = [&](size_t bytes) -> size_t {
    size_t r = off;
    off += (bytes + 255) & ~(size_t)255;
    return r;
  };
  size_t o_xyz    = alloc((size_t)BB * NN * 2 * 4);
  size_t o_fps    = alloc((size_t)BB * SS * 4);
  size_t o_newxyz = alloc((size_t)BB * SS * 2 * 4);
  size_t o_knn    = alloc((size_t)BB * NN * KNBR * 4);
  size_t o_sum    = alloc(256 * 4);
  size_t o_sumsq  = alloc(256 * 4);
  size_t o_scale  = alloc(256 * 4);
  size_t o_shift  = alloc(256 * 4);
  size_t o_w1     = alloc((size_t)64 * 96 * 2);
  size_t o_w2     = alloc((size_t)128 * 64 * 2);
  size_t o_w3     = alloc((size_t)256 * 128 * 2);
  size_t o_R1     = alloc((size_t)128 * P * 2);  // Y1 (64xP) then X2 (Px128)
  size_t o_R2     = alloc((size_t)64 * P * 2);   // X1 (Px64)
  size_t o_R3     = alloc((size_t)256 * P * 2);  // X0 (Px96), Y2 (128xP), Y3 (256xP)

  float*    xyz    = (float*)(base + o_xyz);
  int*      fpsix  = (int*)(base + o_fps);
  float*    newxyz = (float*)(base + o_newxyz);
  int*      knn    = (int*)(base + o_knn);
  float*    bsum   = (float*)(base + o_sum);
  float*    bsumsq = (float*)(base + o_sumsq);
  float*    scale  = (float*)(base + o_scale);
  float*    shift  = (float*)(base + o_shift);
  _Float16* w1p    = (_Float16*)(base + o_w1);
  _Float16* w2p    = (_Float16*)(base + o_w2);
  _Float16* w3p    = (_Float16*)(base + o_w3);
  _Float16* X0     = (_Float16*)(base + o_R3);
  _Float16* Y1     = (_Float16*)(base + o_R1);
  _Float16* X1     = (_Float16*)(base + o_R2);
  _Float16* Y2     = (_Float16*)(base + o_R3);
  _Float16* X2     = (_Float16*)(base + o_R1);
  _Float16* Y3     = (_Float16*)(base + o_R3);

  float* out0 = (float*)d_out;              // [B,2,S]
  float* out2 = out0 + (size_t)BB * 2 * SS; // [B,256,S]

  const float invP = 1.0f / (float)P;

  // 1) xyz transpose, FPS, centers, KNN, input assembly
  k_transpose_xyz<<<(BB * NN + 255) / 256, 256, 0, stream>>>(xy, xyz);
  k_fps<<<BB, 256, 0, stream>>>(xyz, fpsix);
  k_new_xyz<<<(BB * SS + 255) / 256, 256, 0, stream>>>(xyz, fpsix, newxyz, out0);
  k_knn<<<(BB * NN + 255) / 256, 256, 0, stream>>>(xyz, knn);
  k_build_x0<<<(P + 255) / 256, 256, 0, stream>>>(xyz, fea, fpsix, knn, newxyz, X0);

  // 2) pack weights to f16 (K padded to multiple of 32 for layer 1)
  k_pack_w<<<(64 * 96 + 255) / 256, 256, 0, stream>>>(W1, w1p, 64, 66, 96);
  k_pack_w<<<(128 * 64 + 255) / 256, 256, 0, stream>>>(W2, w2p, 128, 64, 64);
  k_pack_w<<<(256 * 128 + 255) / 256, 256, 0, stream>>>(W3, w3p, 256, 128, 128);

  // 3) layer 1: 96 -> 64
  {
    int M = 64, K = 96;
    hipMemsetAsync(bsum, 0, 2 * 256 * 4, stream);  // bsum + bsumsq contiguous
    long long waves = (long long)(M >> 4) * (P >> 4);
    int blocks = (int)((waves * 32 + 255) / 256);
    k_gemm_wmma<<<blocks, 256, 0, stream>>>(w1p, X0, b1, Y1, bsum, bsumsq, M, K, P);
    k_bn_finalize<<<1, 256, 0, stream>>>(bsum, bsumsq, g1, be1, scale, shift, M, invP);
    long long tot = (long long)M * P;
    k_bn_relu_tr<<<(unsigned)((tot + 255) / 256), 256, 0, stream>>>(Y1, scale, shift, X1, M, P);
  }
  // 4) layer 2: 64 -> 128
  {
    int M = 128, K = 64;
    hipMemsetAsync(bsum, 0, 2 * 256 * 4, stream);
    long long waves = (long long)(M >> 4) * (P >> 4);
    int blocks = (int)((waves * 32 + 255) / 256);
    k_gemm_wmma<<<blocks, 256, 0, stream>>>(w2p, X1, b2, Y2, bsum, bsumsq, M, K, P);
    k_bn_finalize<<<1, 256, 0, stream>>>(bsum, bsumsq, g2, be2, scale, shift, M, invP);
    long long tot = (long long)M * P;
    k_bn_relu_tr<<<(unsigned)((tot + 255) / 256), 256, 0, stream>>>(Y2, scale, shift, X2, M, P);
  }
  // 5) layer 3: 128 -> 256, then BN+ReLU+max-over-k straight into d_out
  {
    int M = 256, K = 128;
    hipMemsetAsync(bsum, 0, 2 * 256 * 4, stream);
    long long waves = (long long)(M >> 4) * (P >> 4);
    int blocks = (int)((waves * 32 + 255) / 256);
    k_gemm_wmma<<<blocks, 256, 0, stream>>>(w3p, X2, b3, Y3, bsum, bsumsq, M, K, P);
    k_bn_finalize<<<1, 256, 0, stream>>>(bsum, bsumsq, g3, be3, scale, shift, M, invP);
    int tot = BB * M * SS;
    k_bn_relu_maxk<<<(tot + 255) / 256, 256, 0, stream>>>(Y3, scale, shift, out2, M, P);
  }
}
